// OCARHybridConnector_49624052138022
// MI455X (gfx1250) — matrix operand order repaired and checked by reading, not compiled, not run on hardware
//
#include <hip/hip_runtime.h>
#include <hip/hip_bf16.h>
#include <cstdint>
#include <cstddef>

typedef __bf16 bf16;
typedef __bf16 v16bf __attribute__((ext_vector_type(16)));
typedef float  v8f   __attribute__((ext_vector_type(8)));
typedef unsigned int u32x4 __attribute__((ext_vector_type(4)));

struct Frag32B { u32x4 lo, hi; };   // 32 bytes == one 16xbf16 WMMA fragment

// ---- problem constants --------------------------------------------------
#define DENC   1024
#define DLLM   4096
#define NHEAD  16
#define DHEAD  64
#define NQ     4        // K queries
#define NLAY   4        // L folded layers
#define NBAT   8        // B
#define TSEQ   1500
#define DFF    3072
#define LBF    32       // L*B
#define ROWSH  128      // LB*NQ
#define TOUT   375
#define KCONV  12288    // 3*4096

// ---- CDNA5 async-to-LDS support (guarded; falls back to sync pipeline) --
#if defined(__AMDGCN__) && defined(__has_builtin)
#  if __has_builtin(__builtin_amdgcn_global_load_async_to_lds_b128)
#    define HAVE_ASYNC_LDS 1
#  endif
#endif
#ifndef HAVE_ASYNC_LDS
#  define HAVE_ASYNC_LDS 0
#endif

#if HAVE_ASYNC_LDS
#define AS1 __attribute__((address_space(1)))
#define AS3 __attribute__((address_space(3)))
typedef int gv4i __attribute__((vector_size(16)));   // matches builtin's pointee type
__device__ __forceinline__ void async_ld_b128(const void* g, void* l)
{
    void* gv = (void*)g;
    __builtin_amdgcn_global_load_async_to_lds_b128((AS1 gv4i*)gv, (AS3 gv4i*)l, 0, 0);
}
__device__ __forceinline__ void wait_async0()
{
#if __has_builtin(__builtin_amdgcn_s_wait_asynccnt)
    __builtin_amdgcn_s_wait_asynccnt(0);
#else
    asm volatile("s_wait_asynccnt 0" ::: "memory");
#endif
}
#endif

// =========================================================================
// WMMA GEMM:  C[M,N] = A[M,K] * Bt[N,K]^T + bias[N]
// A, Bt bf16 row-major (Bt is the weight stored [N][K]).  f32 accumulate.
// Block tile 64x128, BK=64, 8 waves, wave tile 16x64 (4 accumulators).
// Double-buffered LDS; async-to-LDS staging when available, else a
// register-double-buffered sync pipeline.  One barrier per K-tile.
// Edge tiles handled by clamping source rows (results never stored).
// =========================================================================
template<bool OUT_BF16>
__global__ __launch_bounds__(256)
void gemm_bf16_wmma(const bf16* __restrict__ A, const bf16* __restrict__ Bt,
                    const float* __restrict__ bias, void* __restrict__ Cout,
                    int M, int N, int Kd)
{
    constexpr int BM = 64, BN = 128, BK = 64, LDP = BK + 8; // pad breaks bank conflicts
    __shared__ bf16 lA[2][BM][LDP];
    __shared__ bf16 lB[2][BN][LDP];

    const int tid  = threadIdx.x;
    const int lane = tid & 31;
    const int wave = tid >> 5;
    const int bm = blockIdx.y * BM;
    const int bn = blockIdx.x * BN;
    const int wm = (wave & 3) * 16;   // wave row inside tile
    const int wn = (wave >> 2) * 64;  // wave col inside tile
    const int ml = lane & 15;
    const int g  = lane >> 4;

    v8f acc[4];
    {
        v8f z = {0.f,0.f,0.f,0.f,0.f,0.f,0.f,0.f};
        acc[0] = z; acc[1] = z; acc[2] = z; acc[3] = z;
    }

    // staging map: thread -> row tid>>2, 8-elem col chunk (tid&3)*8 (and +32)
    const int rS = tid >> 2;
    const int cS = (tid & 3) * 8;
    int gmA = bm + rS;      if (gmA > M - 1) gmA = M - 1;
    int gn0 = bn + rS;      if (gn0 > N - 1) gn0 = N - 1;
    int gn1 = bn + rS + 64; if (gn1 > N - 1) gn1 = N - 1;
    const bf16* pA  = A  + (size_t)gmA * Kd + cS;
    const bf16* pB0 = Bt + (size_t)gn0 * Kd + cS;
    const bf16* pB1 = Bt + (size_t)gn1 * Kd + cS;

    auto compute = [&](int p) {
        #pragma unroll
        for (int s = 0; s < 2; ++s) {
            // A fragment (16x32): lane=M, elems 0..7 -> K=g*8.., 8..15 -> K=g*8+16..
            Frag32B fa;
            fa.lo = *(const u32x4*)(&lA[p][wm + ml][s * 32 + g * 8]);
            fa.hi = *(const u32x4*)(&lA[p][wm + ml][s * 32 + g * 8 + 16]);
            v16bf av = __builtin_bit_cast(v16bf, fa);
            #pragma unroll
            for (int j = 0; j < 4; ++j) {
                // B fragment (32x16): lane=N, elems 0..7 -> K=g*8.., 8..15 -> K=g*8+16..
                Frag32B fb;
                const bf16* bp = &lB[p][wn + j * 16 + ml][s * 32 + g * 8];
                fb.lo = *(const u32x4*)bp;
                fb.hi = *(const u32x4*)(bp + 16);
                v16bf bv = __builtin_bit_cast(v16bf, fb);
                acc[j] = __builtin_amdgcn_wmma_f32_16x16x32_bf16(
                             false, av, false, bv, (short)0, acc[j], false, false);
            }
        }
    };

#if HAVE_ASYNC_LDS
    auto stage_async = [&](int k0, int p) {
        async_ld_b128(pA  + k0,      &lA[p][rS][cS]);
        async_ld_b128(pA  + k0 + 32, &lA[p][rS][cS + 32]);
        async_ld_b128(pB0 + k0,      &lB[p][rS][cS]);
        async_ld_b128(pB0 + k0 + 32, &lB[p][rS][cS + 32]);
        async_ld_b128(pB1 + k0,      &lB[p][rS + 64][cS]);
        async_ld_b128(pB1 + k0 + 32, &lB[p][rS + 64][cS + 32]);
    };
    int p = 0;
    stage_async(0, 0);
    wait_async0();
    __syncthreads();
    for (int k0 = 0; k0 < Kd; k0 += BK) {
        if (k0 + BK < Kd) stage_async(k0 + BK, p ^ 1);  // overlap with WMMA
        compute(p);
        wait_async0();
        __syncthreads();
        p ^= 1;
    }
#else
    // sync fallback: register double-buffer, one barrier per K-tile
    u32x4 rA0, rA1, rB0, rB1, rB2, rB3;
    auto load_regs = [&](int k0) {
        rA0 = *(const u32x4*)(pA  + k0);
        rA1 = *(const u32x4*)(pA  + k0 + 32);
        rB0 = *(const u32x4*)(pB0 + k0);
        rB1 = *(const u32x4*)(pB0 + k0 + 32);
        rB2 = *(const u32x4*)(pB1 + k0);
        rB3 = *(const u32x4*)(pB1 + k0 + 32);
    };
    auto store_regs = [&](int p) {
        *(u32x4*)(&lA[p][rS][cS])           = rA0;
        *(u32x4*)(&lA[p][rS][cS + 32])      = rA1;
        *(u32x4*)(&lB[p][rS][cS])           = rB0;
        *(u32x4*)(&lB[p][rS][cS + 32])      = rB1;
        *(u32x4*)(&lB[p][rS + 64][cS])      = rB2;
        *(u32x4*)(&lB[p][rS + 64][cS + 32]) = rB3;
    };
    int p = 0;
    load_regs(0);
    for (int k0 = 0; k0 < Kd; k0 += BK) {
        store_regs(p);
        __syncthreads();
        if (k0 + BK < Kd) {
            load_regs(k0 + BK);                         // overlap with WMMA
            if (k0 + 2 * BK < Kd) {
                __builtin_prefetch(pA  + k0 + 2 * BK, 0, 1);
                __builtin_prefetch(pB0 + k0 + 2 * BK, 0, 1);
                __builtin_prefetch(pB1 + k0 + 2 * BK, 0, 1);
            }
        }
        compute(p);
        p ^= 1;
    }
#endif

    // ---- epilogue: C/D layout -> lane = N, VGPR r -> M = r + 8*g ----
    #pragma unroll
    for (int j = 0; j < 4; ++j) {
        int n = bn + wn + j * 16 + ml;
        if (n >= N) continue;
        float bv = bias ? bias[n] : 0.0f;
        #pragma unroll
        for (int r = 0; r < 8; ++r) {
            int m = bm + wm + r + 8 * g;
            if (m >= M) continue;
            float val = acc[j][r] + bv;
            if (OUT_BF16) ((bf16*)Cout)[(size_t)m * N + n] = (bf16)val;
            else          ((float*)Cout)[(size_t)m * N + n] = val;
        }
    }
}

// =========================================================================
// small helper kernels
// =========================================================================
__global__ void cast_f32_bf16_kernel(const float* __restrict__ s, bf16* __restrict__ d, size_t n)
{
    size_t i = (size_t)blockIdx.x * blockDim.x + threadIdx.x;
    size_t st = (size_t)gridDim.x * blockDim.x;
    for (; i < n; i += st) d[i] = (bf16)s[i];
}

// W f32 [R][C] -> Wt bf16 [C][R]
__global__ __launch_bounds__(256)
void transpose_cast_kernel(const float* __restrict__ W, bf16* __restrict__ Wt, int R, int C)
{
    __shared__ float tile[32][33];
    int c0 = blockIdx.x * 32, r0 = blockIdx.y * 32;
    int tx = threadIdx.x & 31, ty = threadIdx.x >> 5;
    #pragma unroll
    for (int i = 0; i < 32; i += 8) {
        int r = r0 + ty + i, c = c0 + tx;
        tile[ty + i][tx] = (r < R && c < C) ? W[(size_t)r * C + c] : 0.f;
    }
    __syncthreads();
    #pragma unroll
    for (int i = 0; i < 32; i += 8) {
        int c = c0 + ty + i, r = r0 + tx;
        if (c < C && r < R) Wt[(size_t)c * R + r] = (bf16)tile[tx][ty + i];
    }
}

// conv_W f32 (O=4096, I=4096, S=3) -> Wt bf16 [O][S*4096 + I]
__global__ void convw_pack_kernel(const float* __restrict__ W, bf16* __restrict__ Wt)
{
    size_t total = (size_t)DLLM * KCONV;
    size_t i = (size_t)blockIdx.x * blockDim.x + threadIdx.x;
    size_t st = (size_t)gridDim.x * blockDim.x;
    for (; i < total; i += st) {
        int o   = (int)(i / KCONV);
        int rem = (int)(i % KCONV);
        int s   = rem / DLLM;
        int ic  = rem % DLLM;
        Wt[i] = (bf16)W[((size_t)o * DLLM + ic) * 3 + s];
    }
}

// h init: h[(lb*4+k)*1024+d] = queries[l][k][d], lb = l*8+b
__global__ void init_h_kernel(const float* __restrict__ q, float* __restrict__ h)
{
    size_t total = (size_t)ROWSH * DENC;
    size_t i = (size_t)blockIdx.x * blockDim.x + threadIdx.x;
    size_t st = (size_t)gridDim.x * blockDim.x;
    for (; i < total; i += st) {
        int row = (int)(i >> 10), d = (int)(i & 1023);
        int lb = row >> 2, k = row & 3, l = lb >> 3;
        h[i] = q[((size_t)l * NQ + k) * DENC + d];
    }
}

// 4x4 self-attention among queries; 32 blocks (lb) x 64 threads (head*4+qr)
__global__ void sa_attn_kernel(const float* __restrict__ q, const float* __restrict__ k,
                               const float* __restrict__ v, float* __restrict__ ctx)
{
    int lb = blockIdx.x;
    int h = threadIdx.x >> 2, qr = threadIdx.x & 3;
    const float* qrow = q + ((size_t)(lb * 4 + qr)) * DENC + h * DHEAD;
    float s[4]; float mx = -1e30f;
    #pragma unroll
    for (int kr = 0; kr < 4; ++kr) {
        const float* krow = k + ((size_t)(lb * 4 + kr)) * DENC + h * DHEAD;
        float d = 0.f;
        for (int e = 0; e < DHEAD; ++e) d += qrow[e] * krow[e];
        s[kr] = d * 0.125f;
        mx = fmaxf(mx, s[kr]);
    }
    float den = 0.f;
    #pragma unroll
    for (int kr = 0; kr < 4; ++kr) { s[kr] = __expf(s[kr] - mx); den += s[kr]; }
    float inv = 1.0f / den;
    float* crow = ctx + ((size_t)(lb * 4 + qr)) * DENC + h * DHEAD;
    for (int e = 0; e < DHEAD; ++e) {
        float a = 0.f;
        #pragma unroll
        for (int kr = 0; kr < 4; ++kr)
            a += s[kr] * v[((size_t)(lb * 4 + kr)) * DENC + h * DHEAD + e];
        crow[e] = a * inv;
    }
}

// cross-attn scores: blocks = (lb,h), scores[((lb*16+h)*4+qr)*1500+t]
__global__ __launch_bounds__(256)
void ca_scores_kernel(const float* __restrict__ q, const bf16* __restrict__ kf,
                      float* __restrict__ sc)
{
    int lb = blockIdx.x >> 4, h = blockIdx.x & 15;
    __shared__ float qs[NQ * DHEAD];
    if (threadIdx.x < NQ * DHEAD) {
        int qr = threadIdx.x >> 6, e = threadIdx.x & 63;
        qs[threadIdx.x] = q[((size_t)(lb * 4 + qr)) * DENC + h * DHEAD + e];
    }
    __syncthreads();
    for (int idx = threadIdx.x; idx < NQ * TSEQ; idx += 256) {
        int t = idx >> 2, qr = idx & 3;
        const bf16* kr = kf + ((size_t)(lb * TSEQ + t)) * DENC + h * DHEAD;
        float d = 0.f;
        for (int e = 0; e < DHEAD; ++e) d += qs[qr * DHEAD + e] * (float)kr[e];
        sc[(((size_t)(lb * 16 + h)) * 4 + qr) * TSEQ + t] = d * 0.125f;
    }
}

// row softmax over D elements
__global__ __launch_bounds__(256)
void softmax_rows_kernel(float* __restrict__ x, int D)
{
    __shared__ float red[256];
    float* row = x + (size_t)blockIdx.x * D;
    int tid = threadIdx.x;
    float mx = -1e30f;
    for (int i = tid; i < D; i += 256) mx = fmaxf(mx, row[i]);
    red[tid] = mx; __syncthreads();
    for (int s = 128; s > 0; s >>= 1) { if (tid < s) red[tid] = fmaxf(red[tid], red[tid + s]); __syncthreads(); }
    mx = red[0]; __syncthreads();
    float sum = 0.f;
    for (int i = tid; i < D; i += 256) sum += __expf(row[i] - mx);
    red[tid] = sum; __syncthreads();
    for (int s = 128; s > 0; s >>= 1) { if (tid < s) red[tid] += red[tid + s]; __syncthreads(); }
    float inv = 1.0f / red[0];
    for (int i = tid; i < D; i += 256) row[i] = __expf(row[i] - mx) * inv;
}

// cross-attn ctx: blocks (lb,h), thread = (qr,d)
__global__ __launch_bounds__(256)
void ca_ctx_kernel(const float* __restrict__ sc, const bf16* __restrict__ vf,
                   float* __restrict__ ctx)
{
    int lb = blockIdx.x >> 4, h = blockIdx.x & 15;
    int qr = threadIdx.x >> 6, d = threadIdx.x & 63;
    const float* srow = sc + (((size_t)(lb * 16 + h)) * 4 + qr) * TSEQ;
    float a = 0.f;
    for (int t = 0; t < TSEQ; ++t)
        a += srow[t] * (float)vf[((size_t)(lb * TSEQ + t)) * DENC + h * DHEAD + d];
    ctx[((size_t)(lb * 4 + qr)) * DENC + h * DHEAD + d] = a;
}

// LayerNorm over D, optional residual; out may alias res (per-thread element ownership)
__global__ __launch_bounds__(256)
void ln_kernel(const float* __restrict__ x, const float* __restrict__ res,
               const float* __restrict__ g, const float* __restrict__ b,
               float* __restrict__ out, int D, float eps)
{
    __shared__ float s1[256], s2[256];
    int tid = threadIdx.x;
    const float* xr = x + (size_t)blockIdx.x * D;
    const float* rr = res ? res + (size_t)blockIdx.x * D : nullptr;
    float* orow = out + (size_t)blockIdx.x * D;
    float a = 0.f, q = 0.f;
    for (int i = tid; i < D; i += 256) {
        float v = xr[i] + (rr ? rr[i] : 0.f);
        a += v; q += v * v;
    }
    s1[tid] = a; s2[tid] = q; __syncthreads();
    for (int s = 128; s > 0; s >>= 1) { if (tid < s) { s1[tid] += s1[tid + s]; s2[tid] += s2[tid + s]; } __syncthreads(); }
    float mean = s1[0] / D;
    float var  = s2[0] / D - mean * mean;
    float inv  = rsqrtf(var + eps);
    __syncthreads();
    for (int i = tid; i < D; i += 256) {
        float v = xr[i] + (rr ? rr[i] : 0.f);
        orow[i] = (v - mean) * inv * g[i] + b[i];
    }
}

__global__ void gelu_kernel(float* __restrict__ x, size_t n)
{
    size_t i = (size_t)blockIdx.x * blockDim.x + threadIdx.x;
    size_t st = (size_t)gridDim.x * blockDim.x;
    for (; i < n; i += st) {
        float v = x[i];
        x[i] = 0.5f * v * (1.0f + erff(v * 0.70710678118654752f));
    }
}

// per-(b,k) layer-weighted mix (softmax over L) + LN(1e-5) -> bf16 row for gproj GEMM
__global__ __launch_bounds__(256)
void layer_mix_kernel(const float* __restrict__ h, const float* __restrict__ lw,
                      const float* __restrict__ g, const float* __restrict__ b,
                      bf16* __restrict__ out)
{
    __shared__ float mix[DENC];
    __shared__ float s1[256], s2[256];
    int row = blockIdx.x;                  // b*4 + k
    int bb = row >> 2, kk = row & 3;
    int tid = threadIdx.x;
    float w[NLAY]; float mx = -1e30f;
    #pragma unroll
    for (int l = 0; l < NLAY; ++l) { w[l] = lw[kk * NLAY + l]; mx = fmaxf(mx, w[l]); }
    float den = 0.f;
    #pragma unroll
    for (int l = 0; l < NLAY; ++l) { w[l] = __expf(w[l] - mx); den += w[l]; }
    float inv = 1.0f / den;
    float a = 0.f, q = 0.f;
    for (int d = tid; d < DENC; d += 256) {
        float m = 0.f;
        #pragma unroll
        for (int l = 0; l < NLAY; ++l)
            m += w[l] * inv * h[(((size_t)(l * NBAT + bb)) * NQ + kk) * DENC + d];
        mix[d] = m; a += m; q += m * m;
    }
    s1[tid] = a; s2[tid] = q; __syncthreads();
    for (int s = 128; s > 0; s >>= 1) { if (tid < s) { s1[tid] += s1[tid + s]; s2[tid] += s2[tid + s]; } __syncthreads(); }
    float mean = s1[0] / DENC;
    float var  = s2[0] / DENC - mean * mean;
    float rinv = rsqrtf(var + 1e-5f);
    __syncthreads();
    for (int d = tid; d < DENC; d += 256)
        out[(size_t)row * DENC + d] = (bf16)((mix[d] - mean) * rinv * g[d] + b[d]);
}

// im2col for stride-4 pad-1 k=3 conv: Acol[(b*375+tp)][s*4096+i]
__global__ void im2col_kernel(const bf16* __restrict__ lf, bf16* __restrict__ out)
{
    size_t total = (size_t)NBAT * TOUT * KCONV;
    size_t i = (size_t)blockIdx.x * blockDim.x + threadIdx.x;
    size_t st = (size_t)gridDim.x * blockDim.x;
    for (; i < total; i += st) {
        int row = (int)(i / KCONV);
        int col = (int)(i % KCONV);
        int bb = row / TOUT, tp = row % TOUT;
        int s = col / DLLM, ic = col % DLLM;
        int tin = tp * 4 - 1 + s;
        bf16 v = (bf16)0.f;
        if (tin >= 0 && tin < TSEQ)
            v = lf[((size_t)(bb * TSEQ + tin)) * DLLM + ic];
        out[i] = v;
    }
}

// =========================================================================
// host orchestration
// =========================================================================
extern "C" void kernel_launch(void* const* d_in, const int* in_sizes, int n_in,
                              void* d_out, int out_size, void* d_ws, size_t ws_size,
                              hipStream_t stream)
{
    (void)in_sizes; (void)n_in; (void)out_size; (void)ws_size;

    // ---- inputs (JAX pytree order: alphabetical keys, 'W' < 'b') ----
    const float* target_states = (const float*)d_in[0];
    const float* last_hidden   = (const float*)d_in[1];
    const float* conv_W   = (const float*)d_in[2];
    const float* conv_b   = (const float*)d_in[3];
    const float* gln_b    = (const float*)d_in[4];
    const float* gln_g    = (const float*)d_in[5];
    const float* gproj_W  = (const float*)d_in[6];
    const float* gproj_b  = (const float*)d_in[7];
    const float* layer_w  = (const float*)d_in[8];
    const float* lin_W    = (const float*)d_in[9];
    const float* lin_b    = (const float*)d_in[10];
    const float* lln_b    = (const float*)d_in[11];
    const float* lln_g    = (const float*)d_in[12];
    const float* ca_k_W = (const float*)d_in[13];
    const float* ca_k_b = (const float*)d_in[14];
    const float* ca_ln_b= (const float*)d_in[15];
    const float* ca_ln_g= (const float*)d_in[16];
    const float* ca_o_W = (const float*)d_in[17];
    const float* ca_o_b = (const float*)d_in[18];
    const float* ca_q_W = (const float*)d_in[19];
    const float* ca_q_b = (const float*)d_in[20];
    const float* ca_v_W = (const float*)d_in[21];
    const float* ca_v_b = (const float*)d_in[22];
    const float* ff_W1  = (const float*)d_in[23];
    const float* ff_W2  = (const float*)d_in[24];
    const float* ff_b1  = (const float*)d_in[25];
    const float* ff_b2  = (const float*)d_in[26];
    const float* ff_ln_b= (const float*)d_in[27];
    const float* ff_ln_g= (const float*)d_in[28];
    const float* sa_k_W = (const float*)d_in[29];
    const float* sa_k_b = (const float*)d_in[30];
    const float* sa_ln_b= (const float*)d_in[31];
    const float* sa_ln_g= (const float*)d_in[32];
    const float* sa_o_W = (const float*)d_in[33];
    const float* sa_o_b = (const float*)d_in[34];
    const float* sa_q_W = (const float*)d_in[35];
    const float* sa_q_b = (const float*)d_in[36];
    const float* sa_v_W = (const float*)d_in[37];
    const float* sa_v_b = (const float*)d_in[38];
    const float* queries= (const float*)d_in[39];

    // ---- workspace layout ----
    char* base = (char*)d_ws;
    size_t off = 0;
    auto take = [&](size_t bytes) -> char* {
        char* p = base + off;
        off = (off + bytes + 255) & ~(size_t)255;
        return p;
    };
    const size_t WATT = (size_t)DENC * DENC * 2;      // 1024x1024 bf16
    const size_t WFF  = (size_t)DENC * DFF  * 2;

    bf16 *saqW[2], *sakW[2], *savW[2], *saoW[2];
    bf16 *caqW[2], *cakW[2], *cavW[2], *caoW[2];
    bf16 *f1W[2], *f2W[2];
    for (int l = 0; l < 2; ++l) {
        saqW[l] = (bf16*)take(WATT); sakW[l] = (bf16*)take(WATT);
        savW[l] = (bf16*)take(WATT); saoW[l] = (bf16*)take(WATT);
        caqW[l] = (bf16*)take(WATT); cakW[l] = (bf16*)take(WATT);
        cavW[l] = (bf16*)take(WATT); caoW[l] = (bf16*)take(WATT);
        f1W[l]  = (bf16*)take(WFF);  f2W[l]  = (bf16*)take(WFF);
    }
    bf16* gprojWt = (bf16*)take((size_t)DLLM * DENC * 2);
    bf16* linWt   = (bf16*)take((size_t)DLLM * DENC * 2);
    bf16* convWt  = (bf16*)take((size_t)DLLM * KCONV * 2);

    bf16* encB  = (bf16*)take((size_t)LBF * TSEQ * DENC * 2);   // 98.3 MB
    bf16* lhB   = (bf16*)take((size_t)NBAT * TSEQ * DENC * 2);  // 24.6 MB
    float* h    = (float*)take((size_t)ROWSH * DENC * 4);
    float* bufQ = (float*)take((size_t)ROWSH * DENC * 4);
    float* bufK = (float*)take((size_t)ROWSH * DENC * 4);
    float* bufV = (float*)take((size_t)ROWSH * DENC * 4);
    float* ctx  = (float*)take((size_t)ROWSH * DENC * 4);
    float* ff1  = (float*)take((size_t)ROWSH * DFF  * 4);
    bf16* hb    = (bf16*)take((size_t)ROWSH * DENC * 2);
    bf16* ctxb  = (bf16*)take((size_t)ROWSH * DENC * 2);
    bf16* ff1b  = (bf16*)take((size_t)ROWSH * DFF  * 2);
    bf16* Agp   = (bf16*)take((size_t)LBF * DENC * 2);
    bf16* kfull = (bf16*)take((size_t)LBF * TSEQ * DENC * 2);   // 98.3 MB
    bf16* vfull = (bf16*)take((size_t)LBF * TSEQ * DENC * 2);   // 98.3 MB
    float* scores = (float*)take((size_t)LBF * NHEAD * NQ * TSEQ * 4);

    // aliases (used only after the global branch is done)
    bf16*  lf      = kfull;          // [12000][4096] bf16 = 98.3 MB
    bf16*  Acol    = vfull;          // [3000][12288] bf16 = 73.7 MB
    float* convout = (float*)encB;   // [3000][4096]  f32  = 49.2 MB

    float* gt_out = (float*)d_out;                    // [8][4][4096]
    float* lt_out = (float*)d_out + (size_t)NBAT * NQ * DLLM;

    auto gemm_f = [&](const bf16* A, const bf16* Bt, const float* bias, float* C,
                      int M, int N, int Kd) {
        dim3 grid((N + 127) / 128, (M + 63) / 64);
        gemm_bf16_wmma<false><<<grid, 256, 0, stream>>>(A, Bt, bias, (void*)C, M, N, Kd);
    };
    auto gemm_b = [&](const bf16* A, const bf16* Bt, const float* bias, bf16* C,
                      int M, int N, int Kd) {
        dim3 grid((N + 127) / 128, (M + 63) / 64);
        gemm_bf16_wmma<true><<<grid, 256, 0, stream>>>(A, Bt, bias, (void*)C, M, N, Kd);
    };
    auto cast = [&](const float* s, bf16* d, size_t n) {
        int blocks = (int)((n + 255) / 256);
        if (blocks > 131072) blocks = 131072;
        cast_f32_bf16_kernel<<<blocks, 256, 0, stream>>>(s, d, n);
    };

    // ---- weight prep: transpose+cast to bf16 [N][K] ----
    {
        dim3 tgA((DENC + 31) / 32, (DENC + 31) / 32);
        for (int l = 0; l < 2; ++l) {
            size_t o = (size_t)l * DENC * DENC;
            transpose_cast_kernel<<<tgA, 256, 0, stream>>>(sa_q_W + o, saqW[l], DENC, DENC);
            transpose_cast_kernel<<<tgA, 256, 0, stream>>>(sa_k_W + o, sakW[l], DENC, DENC);
            transpose_cast_kernel<<<tgA, 256, 0, stream>>>(sa_v_W + o, savW[l], DENC, DENC);
            transpose_cast_kernel<<<tgA, 256, 0, stream>>>(sa_o_W + o, saoW[l], DENC, DENC);
            transpose_cast_kernel<<<tgA, 256, 0, stream>>>(ca_q_W + o, caqW[l], DENC, DENC);
            transpose_cast_kernel<<<tgA, 256, 0, stream>>>(ca_k_W + o, cakW[l], DENC, DENC);
            transpose_cast_kernel<<<tgA, 256, 0, stream>>>(ca_v_W + o, cavW[l], DENC, DENC);
            transpose_cast_kernel<<<tgA, 256, 0, stream>>>(ca_o_W + o, caoW[l], DENC, DENC);
            dim3 tg1((DFF + 31) / 32, (DENC + 31) / 32);
            transpose_cast_kernel<<<tg1, 256, 0, stream>>>(ff_W1 + (size_t)l * DENC * DFF, f1W[l], DENC, DFF);
            dim3 tg2((DENC + 31) / 32, (DFF + 31) / 32);
            transpose_cast_kernel<<<tg2, 256, 0, stream>>>(ff_W2 + (size_t)l * DFF * DENC, f2W[l], DFF, DENC);
        }
        dim3 tgP((DLLM + 31) / 32, (DENC + 31) / 32);
        transpose_cast_kernel<<<tgP, 256, 0, stream>>>(gproj_W, gprojWt, DENC, DLLM);
        transpose_cast_kernel<<<tgP, 256, 0, stream>>>(lin_W, linWt, DENC, DLLM);
        convw_pack_kernel<<<131072, 256, 0, stream>>>(conv_W, convWt);
    }

    // ---- activation casts ----
    cast(target_states, encB, (size_t)LBF * TSEQ * DENC);
    cast(last_hidden, lhB, (size_t)NBAT * TSEQ * DENC);
    init_h_kernel<<<512, 256, 0, stream>>>(queries, h);

    // ---- query-former layers ----
    for (int l = 0; l < 2; ++l) {
        const size_t bo = (size_t)l * DENC;
        // self-attention
        cast(h, hb, (size_t)ROWSH * DENC);
        gemm_f(hb, saqW[l], sa_q_b + bo, bufQ, ROWSH, DENC, DENC);
        gemm_f(hb, sakW[l], sa_k_b + bo, bufK, ROWSH, DENC, DENC);
        gemm_f(hb, savW[l], sa_v_b + bo, bufV, ROWSH, DENC, DENC);
        sa_attn_kernel<<<LBF, 64, 0, stream>>>(bufQ, bufK, bufV, ctx);
        cast(ctx, ctxb, (size_t)ROWSH * DENC);
        gemm_f(ctxb, saoW[l], sa_o_b + bo, bufQ, ROWSH, DENC, DENC);
        ln_kernel<<<ROWSH, 256, 0, stream>>>(bufQ, h, sa_ln_g + bo, sa_ln_b + bo, h, DENC, 1e-12f);

        // cross-attention
        cast(h, hb, (size_t)ROWSH * DENC);
        gemm_f(hb, caqW[l], ca_q_b + bo, bufQ, ROWSH, DENC, DENC);
        gemm_b(encB, cakW[l], ca_k_b + bo, kfull, LBF * TSEQ, DENC, DENC);
        gemm_b(encB, cavW[l], ca_v_b + bo, vfull, LBF * TSEQ, DENC, DENC);
        ca_scores_kernel<<<LBF * NHEAD, 256, 0, stream>>>(bufQ, kfull, scores);
        softmax_rows_kernel<<<LBF * NHEAD * NQ, 256, 0, stream>>>(scores, TSEQ);
        ca_ctx_kernel<<<LBF * NHEAD, 256, 0, stream>>>(scores, vfull, ctx);
        cast(ctx, ctxb, (size_t)ROWSH * DENC);
        gemm_f(ctxb, caoW[l], ca_o_b + bo, bufQ, ROWSH, DENC, DENC);
        ln_kernel<<<ROWSH, 256, 0, stream>>>(bufQ, h, ca_ln_g + bo, ca_ln_b + bo, h, DENC, 1e-12f);

        // feed-forward
        cast(h, hb, (size_t)ROWSH * DENC);
        gemm_f(hb, f1W[l], ff_b1 + (size_t)l * DFF, ff1, ROWSH, DFF, DENC);
        gelu_kernel<<<1536, 256, 0, stream>>>(ff1, (size_t)ROWSH * DFF);
        cast(ff1, ff1b, (size_t)ROWSH * DFF);
        gemm_f(ff1b, f2W[l], ff_b2 + bo, bufQ, ROWSH, DENC, DFF);
        ln_kernel<<<ROWSH, 256, 0, stream>>>(bufQ, h, ff_ln_g + bo, ff_ln_b + bo, h, DENC, 1e-12f);
    }

    // ---- global head: layer mix + LN + gproj ----
    layer_mix_kernel<<<NBAT * NQ, 256, 0, stream>>>(h, layer_w, gln_g, gln_b, Agp);
    gemm_f(Agp, gprojWt, gproj_b, gt_out, NBAT * NQ, DLLM, DENC);

    // ---- local branch: linear -> conv (im2col GEMM) -> LN ----
    gemm_b(lhB, linWt, lin_b, lf, NBAT * TSEQ, DLLM, DENC);
    im2col_kernel<<<131072, 256, 0, stream>>>(lf, Acol);
    gemm_f(Acol, convWt, conv_b, convout, NBAT * TOUT, DLLM, KCONV);
    ln_kernel<<<NBAT * TOUT, 256, 0, stream>>>(convout, nullptr, lln_g, lln_b,
                                               lt_out, DLLM, 1e-5f);
}
